// PAMoE_83708912599442
// MI455X (gfx1250) — matrix operand order repaired and checked by reading, not compiled, not run on hardware
//
#include <hip/hip_runtime.h>
#include <hip/hip_bf16.h>
#include <math.h>

// ---------------- problem constants (from reference setup_inputs) -----------
#define B_ 2
#define S_ 2048
#define D_ 1024
#define F_ 2048
#define O_ 1024
#define E_ 8
#define K_CAP 256            // capacity k = S/E
#define NPAIR (B_ * E_)      // 16 (batch, expert) pairs

typedef __bf16 bf16_t;
typedef bf16_t bf16x16 __attribute__((ext_vector_type(16)));
typedef float  v8f     __attribute__((ext_vector_type(8)));

// LDS tile row stride in bf16 elements: 32 data + 8 pad = 80 bytes (16B aligned)
#define LDA 40
#define LDB 40

union FragBF {
  bf16x16 v;
  uint4   q[2];
};

__device__ __forceinline__ float gelu_erf(float x) {
  return 0.5f * x * (1.0f + erff(x * 0.70710678118654752f));
}

// -------- wave-level 64x32 WMMA step over one K=32 slab in LDS --------------
// A layout in LDS: [128 rows][K=32] bf16, row stride LDA
// B layout in LDS: [128 cols(N)][K=32] bf16, row stride LDB (K contiguous)
// Fragment addressing follows CDNA5 ISA 16-bit A/B VGPR layouts (CDNA5 ISA).
__device__ __forceinline__ void wave_mma(const bf16_t* As, const bf16_t* Bs,
                                         int wm, int wn, int half, int l16,
                                         v8f acc[4][2]) {
  FragBF a[4], b[2];
#pragma unroll
  for (int tm = 0; tm < 4; ++tm) {
    int row = wm * 64 + tm * 16 + l16;
    const char* p = (const char*)As + row * (LDA * 2) + half * 16;
    a[tm].q[0] = *(const uint4*)(p);        // K = half*8 + 0..7
    a[tm].q[1] = *(const uint4*)(p + 32);   // K = 16 + half*8 + 0..7
  }
#pragma unroll
  for (int tn = 0; tn < 2; ++tn) {
    int n = wn * 32 + tn * 16 + l16;
    const char* p = (const char*)Bs + n * (LDB * 2) + half * 16;
    b[tn].q[0] = *(const uint4*)(p);        // K = half*8 + 0..7
    b[tn].q[1] = *(const uint4*)(p + 32);   // K = 16 + half*8 + 0..7
  }
#pragma unroll
  for (int tm = 0; tm < 4; ++tm)
#pragma unroll
    for (int tn = 0; tn < 2; ++tn)
      acc[tm][tn] = __builtin_amdgcn_wmma_f32_16x16x32_bf16(
          false, a[tm].v, false, b[tn].v, (short)0, acc[tm][tn], false, false);
}

// ---------------------------- kernel 0: zero output -------------------------
__global__ __launch_bounds__(256) void zero_out_kernel(float* __restrict__ out) {
  size_t i = ((size_t)blockIdx.x * 256 + threadIdx.x) * 4;
  *(float4*)&out[i] = make_float4(0.f, 0.f, 0.f, 0.f);
}

// ---------------------------- kernel 1: gate logits -------------------------
// logits[b,s,e] = sum_d x[b,s,d] * w_gate[d,e]; one wave per token.
__global__ __launch_bounds__(256) void logits_kernel(
    const float* __restrict__ x, const float* __restrict__ wg,
    float* __restrict__ logits) {
  const int tid = threadIdx.x;
  const int lane = tid & 31;
  const int token = blockIdx.x * 8 + (tid >> 5);   // flat over B*S
  float acc[E_] = {};
  for (int d = lane; d < D_; d += 32) {
    float xv = x[(size_t)token * D_ + d];
#pragma unroll
    for (int e = 0; e < E_; ++e) acc[e] += xv * wg[d * E_ + e];
  }
#pragma unroll
  for (int e = 0; e < E_; ++e) {
#pragma unroll
    for (int off = 16; off > 0; off >>= 1)
      acc[e] += __shfl_xor(acc[e], off, 32);
  }
  if (lane == 0) {
#pragma unroll
    for (int e = 0; e < E_; ++e) logits[(size_t)token * E_ + e] = acc[e];
  }
}

// ---------------------------- kernel 2: expert-choice routing ---------------
// Per (b,e): rank each token by logit (ties broken by lower index first, same
// as lax.top_k); rank < k => selected, slot = rank. Gate = softmax_e(logits).
__global__ __launch_bounds__(256) void routing_kernel(
    const float* __restrict__ logits, int* __restrict__ idxb,
    float* __restrict__ gateb) {
  __shared__ float L[S_];
  const int pair = blockIdx.x;
  const int bb = pair >> 3, e = pair & 7;
  const int tid = threadIdx.x;
  for (int i = tid; i < S_; i += 256)
    L[i] = logits[(size_t)(bb * S_ + i) * E_ + e];
  __syncthreads();
  for (int j = 0; j < 8; ++j) {
    int s = j * 256 + tid;
    float v = L[s];
    int rank = 0;
    for (int t2 = 0; t2 < S_; ++t2) {
      float u = L[t2];
      rank += (u > v) || (u == v && t2 < s);
    }
    if (rank < K_CAP) {
      float l8[E_];
      float m = -3.4e38f;
#pragma unroll
      for (int e2 = 0; e2 < E_; ++e2) {
        l8[e2] = logits[(size_t)(bb * S_ + s) * E_ + e2];
        m = fmaxf(m, l8[e2]);
      }
      float sum = 0.f;
#pragma unroll
      for (int e2 = 0; e2 < E_; ++e2) sum += expf(l8[e2] - m);
      idxb[pair * K_CAP + rank] = s;
      gateb[pair * K_CAP + rank] = expf(l8[e] - m) / sum;
    }
  }
}

// ---------------------------- kernel 3: GEMM1 (gathered X @ w1[e]) ----------
// Per block: 128x128 tile of H[pair] = GELU(X_sel @ w1[e] + b1[e]), fp32 out.
__global__ __launch_bounds__(256) void gemm1_kernel(
    const float* __restrict__ x, const float* __restrict__ w1,
    const float* __restrict__ b1, const int* __restrict__ idxb,
    float* __restrict__ H) {
  __shared__ bf16_t As[128 * LDA];
  __shared__ bf16_t Bs[128 * LDB];

  const int blk = blockIdx.x;
  const int ntile = blk & 15;         // F/128 = 16
  const int mtile = (blk >> 4) & 1;   // 256/128 = 2
  const int pair = blk >> 5;
  const int bb = pair >> 3, e = pair & 7;
  const int n0 = ntile * 128;

  const int tid = threadIdx.x;
  const int lane = tid & 31, w = tid >> 5;
  const int wm = w & 1, wn = w >> 1;
  const int half = lane >> 4, l16 = lane & 15;

  // hoist gathered row pointers (4 A-chunks per thread, chunk c = tid + i*256,
  // row = c>>3, 8 float4 chunks per 32-float row)
  const float* xrow[4];
#pragma unroll
  for (int i = 0; i < 4; ++i) {
    int c = tid + i * 256;
    int r = c >> 3;
    int tok = idxb[pair * K_CAP + mtile * 128 + r];
    xrow[i] = &x[((size_t)(bb * S_ + tok)) * D_];
  }

  v8f acc[4][2] = {};

  for (int k0 = 0; k0 < D_; k0 += 32) {
    __syncthreads();
#pragma unroll
    for (int i = 0; i < 4; ++i) {                // A: fp32 -> bf16, row major
      int c = tid + i * 256;
      int r = c >> 3, kq = c & 7;
      const float4 xv = *(const float4*)&xrow[i][k0 + kq * 4];
      bf16_t* dst = &As[r * LDA + kq * 4];
      dst[0] = (bf16_t)xv.x; dst[1] = (bf16_t)xv.y;
      dst[2] = (bf16_t)xv.z; dst[3] = (bf16_t)xv.w;
    }
#pragma unroll
    for (int i = 0; i < 4; ++i) {                // B: fp32 -> bf16, [n][k]
      int c = tid + i * 256;
      int kk = c >> 5, nq = c & 31;
      const float4 wv =
          *(const float4*)&w1[((size_t)e * D_ + k0 + kk) * F_ + n0 + nq * 4];
      Bs[(nq * 4 + 0) * LDB + kk] = (bf16_t)wv.x;
      Bs[(nq * 4 + 1) * LDB + kk] = (bf16_t)wv.y;
      Bs[(nq * 4 + 2) * LDB + kk] = (bf16_t)wv.z;
      Bs[(nq * 4 + 3) * LDB + kk] = (bf16_t)wv.w;
    }
    __syncthreads();
    wave_mma(As, Bs, wm, wn, half, l16, acc);
  }

#pragma unroll
  for (int tm = 0; tm < 4; ++tm)
#pragma unroll
    for (int tn = 0; tn < 2; ++tn)
#pragma unroll
      for (int j = 0; j < 8; ++j) {
        int row = mtile * 128 + wm * 64 + tm * 16 + half * 8 + j;
        int f = n0 + wn * 32 + tn * 16 + l16;
        float v = acc[tm][tn][j] + b1[e * F_ + f];
        H[((size_t)pair * K_CAP + row) * F_ + f] = gelu_erf(v);
      }
}

// ---------------------------- kernel 4: LayerNorm over F --------------------
__global__ __launch_bounds__(256) void ln_kernel(
    const float* __restrict__ H, const float* __restrict__ g_ln,
    const float* __restrict__ b_ln, unsigned short* __restrict__ Hln_u) {
  bf16_t* Hln = (bf16_t*)Hln_u;
  const int pair = blockIdx.x >> 8;
  const int row = blockIdx.x & 255;
  const int e = pair & 7;
  const int tid = threadIdx.x;
  const size_t base = ((size_t)pair * K_CAP + row) * F_;

  __shared__ float rs[256], rq[256];
  float v[8], s = 0.f, q = 0.f;
#pragma unroll
  for (int i = 0; i < 8; ++i) {
    int f = i * 256 + tid;
    v[i] = H[base + f];
    s += v[i];
    q += v[i] * v[i];
  }
  rs[tid] = s; rq[tid] = q;
  __syncthreads();
  for (int off = 128; off > 0; off >>= 1) {
    if (tid < off) { rs[tid] += rs[tid + off]; rq[tid] += rq[tid + off]; }
    __syncthreads();
  }
  const float mean = rs[0] * (1.0f / F_);
  const float var = rq[0] * (1.0f / F_) - mean * mean;
  const float rstd = rsqrtf(var + 1e-5f);
#pragma unroll
  for (int i = 0; i < 8; ++i) {
    int f = i * 256 + tid;
    float hn = (v[i] - mean) * rstd * g_ln[e * F_ + f] + b_ln[e * F_ + f];
    Hln[base + f] = (bf16_t)hn;
  }
}

// ---------------------------- kernel 5: GEMM2 + gated scatter-add -----------
__global__ __launch_bounds__(256) void gemm2_kernel(
    const unsigned short* __restrict__ Hln_u, const float* __restrict__ w2,
    const float* __restrict__ b2, const int* __restrict__ idxb,
    const float* __restrict__ gateb, float* __restrict__ out) {
  const bf16_t* Hln = (const bf16_t*)Hln_u;
  __shared__ bf16_t As[128 * LDA];
  __shared__ bf16_t Bs[128 * LDB];

  const int blk = blockIdx.x;
  const int ntile = blk & 7;          // O/128 = 8
  const int mtile = (blk >> 3) & 1;
  const int pair = blk >> 4;
  const int bb = pair >> 3, e = pair & 7;
  const int n0 = ntile * 128;

  const int tid = threadIdx.x;
  const int lane = tid & 31, w = tid >> 5;
  const int wm = w & 1, wn = w >> 1;
  const int half = lane >> 4, l16 = lane & 15;

  v8f acc[4][2] = {};

  for (int k0 = 0; k0 < F_; k0 += 32) {
    __syncthreads();
#pragma unroll
    for (int i = 0; i < 2; ++i) {                // A: already bf16, b128 copy
      int c = tid + i * 256;
      int r = c >> 2, kq = c & 3;                // 4x 8-elem chunks per row
      const uint4 hv = *(const uint4*)&Hln[((size_t)pair * K_CAP + mtile * 128 + r) * F_ +
                                           k0 + kq * 8];
      *(uint4*)&As[r * LDA + kq * 8] = hv;
    }
#pragma unroll
    for (int i = 0; i < 4; ++i) {                // B: w2 fp32 -> bf16, [n][k]
      int c = tid + i * 256;
      int kk = c >> 5, nq = c & 31;
      const float4 wv =
          *(const float4*)&w2[((size_t)e * F_ + k0 + kk) * O_ + n0 + nq * 4];
      Bs[(nq * 4 + 0) * LDB + kk] = (bf16_t)wv.x;
      Bs[(nq * 4 + 1) * LDB + kk] = (bf16_t)wv.y;
      Bs[(nq * 4 + 2) * LDB + kk] = (bf16_t)wv.z;
      Bs[(nq * 4 + 3) * LDB + kk] = (bf16_t)wv.w;
    }
    __syncthreads();
    wave_mma(As, Bs, wm, wn, half, l16, acc);
  }

#pragma unroll
  for (int tm = 0; tm < 4; ++tm)
#pragma unroll
    for (int tn = 0; tn < 2; ++tn)
#pragma unroll
      for (int j = 0; j < 8; ++j) {
        int row = mtile * 128 + wm * 64 + tm * 16 + half * 8 + j;
        int o = n0 + wn * 32 + tn * 16 + l16;
        int tok = idxb[pair * K_CAP + row];
        float g = gateb[pair * K_CAP + row];
        float y = acc[tm][tn][j] + b2[e * O_ + o];
        atomicAdd(&out[((size_t)(bb * S_ + tok)) * O_ + o], g * y);
      }
}

// ---------------------------- launch ---------------------------------------
extern "C" void kernel_launch(void* const* d_in, const int* in_sizes, int n_in,
                              void* d_out, int out_size, void* d_ws, size_t ws_size,
                              hipStream_t stream) {
  (void)in_sizes; (void)n_in; (void)out_size; (void)ws_size;
  const float* x      = (const float*)d_in[0];
  const float* w_gate = (const float*)d_in[1];
  const float* w1     = (const float*)d_in[2];
  const float* b1     = (const float*)d_in[3];
  const float* g_ln   = (const float*)d_in[4];
  const float* b_ln   = (const float*)d_in[5];
  const float* w2     = (const float*)d_in[6];
  const float* b2     = (const float*)d_in[7];

  float* out = (float*)d_out;                      // [B,S,O]
  float* logits = out + (size_t)B_ * S_ * O_;      // [B,S,E] (2nd output)

  char* ws = (char*)d_ws;
  int*   idxb  = (int*)ws;                                    // 16 KB
  float* gateb = (float*)(ws + 16384);                        // 16 KB
  float* H     = (float*)(ws + 32768);                        // 32 MB fp32
  unsigned short* Hln =
      (unsigned short*)(ws + 32768 + (size_t)NPAIR * K_CAP * F_ * sizeof(float));

  zero_out_kernel<<<(B_ * S_ * O_) / (256 * 4), 256, 0, stream>>>(out);
  logits_kernel<<<(B_ * S_) / 8, 256, 0, stream>>>(x, w_gate, logits);
  routing_kernel<<<NPAIR, 256, 0, stream>>>(logits, idxb, gateb);
  gemm1_kernel<<<NPAIR * 2 * (F_ / 128), 256, 0, stream>>>(x, w1, b1, idxb, H);
  ln_kernel<<<NPAIR * K_CAP, 256, 0, stream>>>(H, g_ln, b_ln, Hln);
  gemm2_kernel<<<NPAIR * 2 * (O_ / 128), 256, 0, stream>>>(Hln, w2, b2, idxb, gateb, out);
}
